// AxialAttentionBlock_56650618634908
// MI455X (gfx1250) — compile-verified, matrix operand order checked by reading, not run on hardware
//
#include <hip/hip_runtime.h>

typedef __attribute__((ext_vector_type(16))) _Float16 v16h;
typedef __attribute__((ext_vector_type(8)))  _Float16 v8h;
typedef __attribute__((ext_vector_type(8)))  float    v8f;

#define NH 8
#define SQ 128        // sequence length
#define CH 512        // channels
#define DH 64         // head dim
#define HW (128*128)

// ---- LDS layout (units: halfs). Row strides padded for conflict-free b128. ----
#define XS 520                         // X row stride (512 cols)
#define OS 520                         // O_all row stride (512 cols)
#define QS 72                          // Q/K row stride (64 cols)
#define VS 136                         // V^T row stride (128 cols)
#define PS 136                         // P row stride (128 cols)

#define X_OFF 0
#define O_OFF (X_OFF + 128 * XS)       // 66560
#define Q_OFF (O_OFF + 128 * OS)       // 133120
#define K_OFF (Q_OFF + 128 * QS)       // 142336
#define V_OFF (K_OFF + 128 * QS)       // 151552
#define P_OFF Q_OFF                    // P (17408 h) overlaps dead Q+K (18432 h)
#define LDS_HALFS (V_OFF + 64 * VS)    // 160256 halfs = 320512 bytes <= 320KB

// 16-bit A/B WMMA fragment: each lane takes two contiguous 16B chunks at
// (k + 8*(lane>=16)) and (+16). Caller pre-offsets p by row*stride + k + half8.
static __device__ __forceinline__ v16h load_frag(const _Float16* p) {
    v8h a = *(const v8h*)(p);
    v8h b = *(const v8h*)(p + 16);
    v16h r;
#pragma unroll
    for (int i = 0; i < 8; ++i) { r[i] = a[i]; r[i + 8] = b[i]; }
    return r;
}

static __device__ __forceinline__ v8f wmma16(v16h a, v16h b, v8f c) {
    return __builtin_amdgcn_wmma_f32_16x16x32_f16(false, a, false, b, (short)0, c,
                                                  false, false);
}

static __device__ __forceinline__ v8f vzero() {
    v8f z;
#pragma unroll
    for (int i = 0; i < 8; ++i) z[i] = 0.0f;
    return z;
}

__global__ void cvt_w_kernel(const float* src, _Float16* dst, int n) {
    for (int i = blockIdx.x * blockDim.x + threadIdx.x; i < n;
         i += gridDim.x * blockDim.x)
        dst[i] = (_Float16)src[i];
}

// QKV projection, register-blocked: this wave computes a 32x96 block of the
// 128x192 (Q|K|V) concat for head h. CB selects column half (compile-time so
// the Q/K/Vt store targets resolve statically). 2 A-frags x 6 B-frags ->
// 12 WMMAs per k-step with only 6 global fragment loads.
template <int CB>
static __device__ __forceinline__ void qkv_head(
    const _Float16* Wq, const _Float16* Wk, const _Float16* Wv,
    _Float16* lds, int h, int R0, int ln15, int half8)
{
    v8f acc[2][6];
#pragma unroll
    for (int i = 0; i < 2; ++i)
#pragma unroll
        for (int j = 0; j < 6; ++j) acc[i][j] = vzero();

    const _Float16* wp[6];
#pragma unroll
    for (int j = 0; j < 6; ++j) {
        const int ct = CB * 6 + j;                 // concat col tile 0..11
        const int mat = ct >> 2, tt = ct & 3;      // matrix, tile within matrix
        const _Float16* W = (mat == 0) ? Wq : (mat == 1) ? Wk : Wv;
        wp[j] = W + (h * DH + tt * 16 + ln15) * CH;
    }

    for (int k = 0; k < CH; k += 32) {
        v16h a0 = load_frag(&lds[X_OFF + (R0 + ln15) * XS + k + half8]);
        v16h a1 = load_frag(&lds[X_OFF + (R0 + 16 + ln15) * XS + k + half8]);
#pragma unroll
        for (int j = 0; j < 6; ++j) {
            v16h bb = load_frag(wp[j] + k + half8);
            acc[0][j] = wmma16(a0, bb, acc[0][j]);
            acc[1][j] = wmma16(a1, bb, acc[1][j]);
        }
    }

#pragma unroll
    for (int j = 0; j < 6; ++j) {
        const int ct = CB * 6 + j;
        const int mat = ct >> 2, tt = ct & 3;
        const int n = tt * 16 + ln15;              // head-local d
#pragma unroll
        for (int i = 0; i < 2; ++i) {
            const int mrow = R0 + i * 16 + half8;
            if (mat == 2) {                        // V stored transposed Vt[d][m]
                v8h pk;
#pragma unroll
                for (int r = 0; r < 8; ++r) pk[r] = (_Float16)acc[i][j][r];
                *(v8h*)&lds[V_OFF + n * VS + mrow] = pk;
            } else {
                const int base = (mat == 0) ? Q_OFF : K_OFF;
#pragma unroll
                for (int r = 0; r < 8; ++r)
                    lds[base + (mrow + r) * QS + n] = (_Float16)acc[i][j][r];
            }
        }
    }
}

// One workgroup (8 waves) = one full sequence.
// pass==0: sequence along H, fixed w=fx, out = addsrc(xs) + o
// pass==1: sequence along W, fixed h=fx, out = addsrc(out) + o
__global__ __launch_bounds__(256) void axial_attn_kernel(
    const float* __restrict__ xs, const _Float16* __restrict__ wts,
    const float* addsrc, float* out, int pass)
{
    extern __shared__ _Float16 lds[];
    const int tid   = threadIdx.x;
    const int lane  = tid & 31;
    const int wv    = tid >> 5;
    const int ln15  = lane & 15;
    const int half8 = (lane >> 4) << 3;    // 0 or 8
    const int mb    = wv * 16;             // attention row block (softmax mapping)

    const int b  = blockIdx.x >> 7;
    const int fx = blockIdx.x & 127;
    const int m_stride  = pass ? 1 : 128;
    const long seq_base = (long)b * ((long)CH * HW) + (pass ? fx * 128 : fx);

    const _Float16* Wq = wts;
    const _Float16* Wk = wts + CH * CH;
    const _Float16* Wv = wts + 2 * CH * CH;
    const _Float16* Wo = wts + 3 * CH * CH;

    // ---- gather X (S x C) -> f16 LDS ----
    if (pass) {
        // m contiguous in memory: coalesced float4 loads
        for (int idx = tid; idx < SQ * CH / 4; idx += 256) {
            int m4 = (idx & 31) * 4;
            int c  = idx >> 5;
            float4 v = *(const float4*)(xs + seq_base + (long)c * HW + m4);
            lds[X_OFF + (m4 + 0) * XS + c] = (_Float16)v.x;
            lds[X_OFF + (m4 + 1) * XS + c] = (_Float16)v.y;
            lds[X_OFF + (m4 + 2) * XS + c] = (_Float16)v.z;
            lds[X_OFF + (m4 + 3) * XS + c] = (_Float16)v.w;
        }
    } else {
        for (int idx = tid; idx < SQ * CH; idx += 256) {
            int m = idx & (SQ - 1);
            int c = idx >> 7;
            lds[X_OFF + m * XS + c] =
                (_Float16)xs[seq_base + (long)c * HW + (long)m * 128];
        }
    }
    __syncthreads();

    const int qkvR0 = (wv >> 1) * 32;      // 32-row pair for projection blocking
    const int qkvCB = wv & 1;

    for (int h = 0; h < NH; ++h) {
        // ---- Q,K,V = X @ W_h^T (register-blocked 32x96 per wave) ----
        if (qkvCB == 0)
            qkv_head<0>(Wq, Wk, Wv, lds, h, qkvR0, ln15, half8);
        else
            qkv_head<1>(Wq, Wk, Wv, lds, h, qkvR0, ln15, half8);
        __syncthreads();

        // ---- scores = Q @ K^T (wave: rows mb..mb+15, all 128 cols) ----
        v8f sc[8];
#pragma unroll
        for (int t = 0; t < 8; ++t) sc[t] = vzero();
#pragma unroll
        for (int k = 0; k < DH; k += 32) {
            v16h a = load_frag(&lds[Q_OFF + (mb + ln15) * QS + k + half8]);
#pragma unroll
            for (int t = 0; t < 8; ++t) {
                v16h bb = load_frag(&lds[K_OFF + (t * 16 + ln15) * QS + k + half8]);
                sc[t] = wmma16(a, bb, sc[t]);
            }
        }

        // ---- softmax over columns; wave halves hold rows r and r+8 ----
        float rcp[8];
#pragma unroll
        for (int r = 0; r < 8; ++r) {
            float mx = -1e30f;
#pragma unroll
            for (int t = 0; t < 8; ++t) mx = fmaxf(mx, sc[t][r]);
#pragma unroll
            for (int off = 8; off >= 1; off >>= 1)
                mx = fmaxf(mx, __shfl_xor(mx, off, 16));
            float sum = 0.0f;
#pragma unroll
            for (int t = 0; t < 8; ++t) {
                // softmax(s * dh^-0.5): 0.125 * log2(e)
                float e = exp2f((sc[t][r] - mx) * 0.18033688f);
                sc[t][r] = e;
                sum += e;
            }
#pragma unroll
            for (int off = 8; off >= 1; off >>= 1)
                sum += __shfl_xor(sum, off, 16);
            rcp[r] = 1.0f / sum;
        }
        __syncthreads();   // all waves done reading Q/K before P overwrites them

        // ---- store P (unnormalized; 1/sum folded into O) ----
#pragma unroll
        for (int t = 0; t < 8; ++t) {
#pragma unroll
            for (int r = 0; r < 8; ++r)
                lds[P_OFF + (mb + half8 + r) * PS + t * 16 + ln15] =
                    (_Float16)sc[t][r];
        }

        // ---- O = P @ V (B-fragments from Vt rows: k-contiguous) ----
        v8f oa[4];
#pragma unroll
        for (int t = 0; t < 4; ++t) oa[t] = vzero();
#pragma unroll
        for (int k = 0; k < SQ; k += 32) {
            v16h a = load_frag(&lds[P_OFF + (mb + ln15) * PS + k + half8]);
#pragma unroll
            for (int t = 0; t < 4; ++t) {
                v16h bb = load_frag(&lds[V_OFF + (t * 16 + ln15) * VS + k + half8]);
                oa[t] = wmma16(a, bb, oa[t]);
            }
        }
#pragma unroll
        for (int t = 0; t < 4; ++t) {
            int n = h * DH + t * 16 + ln15;        // concat heads along C
#pragma unroll
            for (int r = 0; r < 8; ++r)
                lds[O_OFF + (mb + half8 + r) * OS + n] =
                    (_Float16)(oa[t][r] * rcp[r]);
        }
        __syncthreads();   // before next head reuses Q/K/V/P regions
    }

    // ---- out = addsrc + O_all @ Wo^T : two 32x128 blocks per wave ----
    for (int blk = wv; blk < 16; blk += 8) {
        const int R0 = (blk >> 2) * 32;
        const int N0 = (blk & 3) * 128;
        v8f acc[2][8];
#pragma unroll
        for (int i = 0; i < 2; ++i)
#pragma unroll
            for (int t = 0; t < 8; ++t) acc[i][t] = vzero();

        for (int k = 0; k < CH; k += 32) {
            v16h a0 = load_frag(&lds[O_OFF + (R0 + ln15) * OS + k + half8]);
            v16h a1 = load_frag(&lds[O_OFF + (R0 + 16 + ln15) * OS + k + half8]);
#pragma unroll
            for (int t = 0; t < 8; ++t) {
                v16h bb = load_frag(Wo + (N0 + t * 16 + ln15) * CH + k + half8);
                acc[0][t] = wmma16(a0, bb, acc[0][t]);
                acc[1][t] = wmma16(a1, bb, acc[1][t]);
            }
        }

#pragma unroll
        for (int t = 0; t < 8; ++t) {
            const int n = N0 + t * 16 + ln15;
#pragma unroll
            for (int i = 0; i < 2; ++i) {
                const int m0 = R0 + i * 16 + half8;
                if (m_stride == 1) {               // pass W: m contiguous
                    long g = seq_base + (long)n * HW + m0;
                    float4 s0 = *(const float4*)(addsrc + g);
                    float4 s1 = *(const float4*)(addsrc + g + 4);
                    float4 d0, d1;
                    d0.x = s0.x + acc[i][t][0]; d0.y = s0.y + acc[i][t][1];
                    d0.z = s0.z + acc[i][t][2]; d0.w = s0.w + acc[i][t][3];
                    d1.x = s1.x + acc[i][t][4]; d1.y = s1.y + acc[i][t][5];
                    d1.z = s1.z + acc[i][t][6]; d1.w = s1.w + acc[i][t][7];
                    *(float4*)(out + g)     = d0;
                    *(float4*)(out + g + 4) = d1;
                } else {                           // pass H: stride 128 floats
#pragma unroll
                    for (int r = 0; r < 8; ++r) {
                        long g = seq_base + (long)n * HW + (long)(m0 + r) * 128;
                        out[g] = addsrc[g] + acc[i][t][r];
                    }
                }
            }
        }
    }
}

extern "C" void kernel_launch(void* const* d_in, const int* in_sizes, int n_in,
                              void* d_out, int out_size, void* d_ws, size_t ws_size,
                              hipStream_t stream) {
    (void)in_sizes; (void)n_in; (void)out_size; (void)ws_size;
    const float* xs = (const float*)d_in[0];
    _Float16* wbuf = (_Float16*)d_ws;      // 8 x 512x512 f16 = 4 MB

    // Convert weights (order: Wq_h, Wk_h, Wv_h, Wo_h, Wq_w, Wk_w, Wv_w, Wo_w)
    for (int i = 0; i < 8; ++i)
        cvt_w_kernel<<<512, 256, 0, stream>>>((const float*)d_in[1 + i],
                                              wbuf + (size_t)i * CH * CH, CH * CH);

    float* out = (float*)d_out;
    const size_t lds_bytes = (size_t)LDS_HALFS * sizeof(_Float16);

    // Pass H: sequences along H (one block per (b,w)); out = xs + oh
    axial_attn_kernel<<<1024, 256, lds_bytes, stream>>>(xs, wbuf, xs, out, 0);
    // Pass W: sequences along W (one block per (b,h)); out += ow
    axial_attn_kernel<<<1024, 256, lds_bytes, stream>>>(xs, wbuf + 4 * CH * CH,
                                                        out, out, 1);
}
